// FastVCompressor_4209067950092
// MI455X (gfx1250) — matrix operand order repaired and checked by reading, not compiled, not run on hardware
//
#include <hip/hip_runtime.h>
#include <hip/hip_bf16.h>

// ---------------------------------------------------------------------------
// Problem constants (from reference): B=4, S=4096, H=1024, C=256
// ---------------------------------------------------------------------------
#define HDIM   1024
#define NCODE  256
#define NROWS  16384                       // B*S rows per tensor
#define NTILES_PER_TENSOR (NROWS / 16)     // 1024 row-tiles of 16
#define KT_COUNT (HDIM / 32)               // 32 k-tiles of 32
#define NT_COUNT (NCODE / 16)              // 16 n-tiles of 16

#define STAGE_KT  4                        // k-tiles staged per LDS buffer
#define NSTAGES   (KT_COUNT / STAGE_KT)    // 8
#define SLICE_USH (NT_COUNT * 32 * 16)     // ushorts per k-tile slice (16KB)
#define STAGE_USH (STAGE_KT * SLICE_USH)   // ushorts per stage (64KB)

typedef __attribute__((ext_vector_type(16))) __bf16          v16bf;
typedef __attribute__((ext_vector_type(16))) unsigned short  v16us;
typedef __attribute__((ext_vector_type(8)))  float           v8f;

union FragB16 {
    v16us u;
    v16bf b;
};

__device__ __forceinline__ unsigned short f32_to_bf16_rne(float f) {
    unsigned int u = __float_as_uint(f);
    unsigned int r = u + 0x7FFFu + ((u >> 16) & 1u);   // round-to-nearest-even
    return (unsigned short)(r >> 16);
}

// 16-byte async copy global -> LDS (ASYNCcnt-tracked, bypasses VGPRs)
__device__ __forceinline__ void async_copy16(const void* gsrc, unsigned lds_off) {
    unsigned long long ga = (unsigned long long)gsrc;
    asm volatile("global_load_async_to_lds_b128 %0, %1, off"
                 :: "v"(lds_off), "v"(ga) : "memory");
}

// ---------------------------------------------------------------------------
// Kernel 1: per-code squared norms ||c||^2  (sparsity mask: ||c||^2 > 0.01)
// ---------------------------------------------------------------------------
__global__ __launch_bounds__(256) void csq_kernel(const float* __restrict__ cb,
                                                  float* __restrict__ c_sq) {
    int n = blockIdx.x * blockDim.x + threadIdx.x;
    if (n >= NCODE) return;
    const float4* row = (const float4*)(cb + (size_t)n * HDIM);
    float s = 0.0f;
    for (int i = 0; i < HDIM / 4; ++i) {
        float4 v = row[i];
        s += v.x * v.x + v.y * v.y + v.z * v.z + v.w * v.w;
    }
    c_sq[n] = s;
}

// ---------------------------------------------------------------------------
// Kernel 2: pack codebook into WMMA B-fragment layout (bf16), kt-major so
// each k-slice is a contiguous 16KB block for the async LDS staging:
//   dst index = ((kt*16 + nt)*32 + lane)*16 + e
//   N = nt*16 + (lane & 15),  K = kt*32 + 8*(lane>>4) + (e&7) + 16*(e>>3)
// ---------------------------------------------------------------------------
__global__ __launch_bounds__(256) void pack_kernel(const float* __restrict__ cb,
                                                   unsigned short* __restrict__ cb_frag) {
    unsigned tid  = blockIdx.x * blockDim.x + threadIdx.x;   // 0 .. 262143
    unsigned e    = tid & 15u;
    unsigned lane = (tid >> 4) & 31u;
    unsigned nt   = (tid >> 9) & 15u;
    unsigned kt   = tid >> 13;
    unsigned n = nt * 16u + (lane & 15u);
    unsigned k = kt * 32u + ((lane >> 4) << 3) + (e & 7u) + ((e >> 3) << 4);
    cb_frag[tid] = f32_to_bf16_rne(cb[(size_t)n * HDIM + k]);
}

// ---------------------------------------------------------------------------
// Kernel 3: main VQ kernel. 8 waves/WG, one 16-row tile per wave. The packed
// codebook (shared by all waves) is staged through a double-buffered LDS
// pipeline with async global->LDS copies; WMMAs consume B via ds_load_b128.
// ---------------------------------------------------------------------------
__global__ __launch_bounds__(256) void vq_main(const float* __restrict__ keys,
                                               const float* __restrict__ values,
                                               const float* __restrict__ cb,
                                               const float* __restrict__ c_sq,
                                               const unsigned short* __restrict__ cb_frag,
                                               float* __restrict__ out) {
    __shared__ __attribute__((aligned(32))) unsigned short lds_b[2][STAGE_USH];

    const int tid  = threadIdx.x;
    const int lane = tid & 31;
    const int wave = tid >> 5;
    const int tile = blockIdx.x * 8 + wave;          // 0 .. 2047
    const int tensor = tile >> 10;                   // 0 = keys, 1 = values
    const int row0 = (tile & (NTILES_PER_TENSOR - 1)) << 4;

    const float* xbase = tensor ? values : keys;
    const float* xrow = xbase + (size_t)(row0 + (lane & 15)) * HDIM;
    float* outbase = out + (size_t)tensor * NROWS * HDIM;

    const unsigned lbase[2] = { (unsigned)(uintptr_t)(&lds_b[0][0]),
                                (unsigned)(uintptr_t)(&lds_b[1][0]) };

    v8f acc[NT_COUNT];
    v8f z = {};
#pragma unroll
    for (int nt = 0; nt < NT_COUNT; ++nt) acc[nt] = z;

    // 16-bit A 16x32 layout: lanes 0-15 K in {0..7,16..23}, lanes 16-31 +8
    const int koffA = (lane >> 4) << 3;              // 0 or 8

    // ---- prologue: kick off stage 0 copy (16 x 16B per thread = 64KB/WG) --
#pragma unroll
    for (int i = 0; i < 16; ++i) {
        unsigned off = (unsigned)(i * 256 + tid) * 16u;   // bytes within stage
        async_copy16((const char*)cb_frag + off, lbase[0] + off);
    }

    for (int s = 0; s < NSTAGES; ++s) {
        // issue next stage into the other buffer while this one drains
        if (s + 1 < NSTAGES) {
            const char* gs = (const char*)cb_frag +
                             (size_t)(s + 1) * (STAGE_USH * 2);
            unsigned lb = lbase[(s + 1) & 1];
#pragma unroll
            for (int i = 0; i < 16; ++i) {
                unsigned off = (unsigned)(i * 256 + tid) * 16u;
                async_copy16(gs + off, lb + off);
            }
            asm volatile("s_wait_asynccnt 16" ::: "memory"); // stage s done
        } else {
            asm volatile("s_wait_asynccnt 0" ::: "memory");
        }
        __syncthreads();   // all waves' copies for stage s are in LDS

        const unsigned short* bstage = &lds_b[s & 1][0];

#pragma unroll
        for (int ktl = 0; ktl < STAGE_KT; ++ktl) {
            const int kt = s * STAGE_KT + ktl;

            // ---- A fragment: two aligned float4 pairs per lane, native cvt
            const float4* p0 = (const float4*)(xrow + kt * 32 + koffA);
            float4 f0 = p0[0];
            float4 f1 = p0[1];
            const float4* p1 = (const float4*)(xrow + kt * 32 + koffA + 16);
            float4 f2 = p1[0];
            float4 f3 = p1[1];
            float fa[16] = { f0.x, f0.y, f0.z, f0.w,  f1.x, f1.y, f1.z, f1.w,
                             f2.x, f2.y, f2.z, f2.w,  f3.x, f3.y, f3.z, f3.w };
            v16bf a;
#pragma unroll
            for (int e = 0; e < 16; ++e) a[e] = (__bf16)fa[e];

            // ---- 16 B fragments from LDS + WMMA --------------------------
#pragma unroll
            for (int nt = 0; nt < NT_COUNT; ++nt) {
                FragB16 bf;
                bf.u = *(const v16us*)(bstage + ktl * SLICE_USH +
                                       nt * 512 + lane * 16);
                acc[nt] = __builtin_amdgcn_wmma_f32_16x16x32_bf16(
                    /*neg_a=*/false, a, /*neg_b=*/false, bf.b,
                    /*c_mod=*/(short)0, acc[nt],
                    /*reuse_a=*/false, /*reuse_b=*/false);
            }
        }
        __syncthreads();   // everyone done reading buf[s&1] before refill
    }

    // ---- per-lane argmin of (||c||^2 - 2 x.c) over the nt chunks ----------
    // C/D layout: lane group (lane>>4) holds rows M = j + 8*(lane>>4),
    // column N = nt*16 + (lane & 15).
    const int nlocal = lane & 15;
    float bestv[8];
    int   besti[8];
#pragma unroll
    for (int j = 0; j < 8; ++j) { bestv[j] = 3.402823466e38f; besti[j] = 0; }

#pragma unroll
    for (int nt = 0; nt < NT_COUNT; ++nt) {
        int n = nt * 16 + nlocal;
        float csq = c_sq[n];
#pragma unroll
        for (int j = 0; j < 8; ++j) {
            float s = csq - 2.0f * acc[nt][j];
            bool take = (s < bestv[j]) || (s == bestv[j] && n < besti[j]);
            bestv[j] = take ? s : bestv[j];
            besti[j] = take ? n : besti[j];
        }
    }

    // cross-lane reduction within each 16-lane half (xor 1,2,4,8 stays in half)
#pragma unroll
    for (int off = 8; off >= 1; off >>= 1) {
#pragma unroll
        for (int j = 0; j < 8; ++j) {
            float ov = __shfl_xor(bestv[j], off, 32);
            int   oi = __shfl_xor(besti[j], off, 32);
            bool take = (ov < bestv[j]) || (ov == bestv[j] && oi < besti[j]);
            bestv[j] = take ? ov : bestv[j];
            besti[j] = take ? oi : besti[j];
        }
    }

    // ---- gather selected codebook rows, apply sparsity mask, store --------
#pragma unroll 1
    for (int m = 0; m < 16; ++m) {
        int j = m & 7;
        int idx = __shfl(besti[j], (m < 8) ? 0 : 16, 32);
        float scale = (c_sq[idx] > 0.01f) ? 1.0f : 0.0f;   // ||c|| > 0.1
        const float4* crow = (const float4*)(cb + (size_t)idx * HDIM);
        float4* orow = (float4*)(outbase + (size_t)(row0 + m) * HDIM);
#pragma unroll
        for (int i = 0; i < HDIM / 128; ++i) {             // 8 iters
            float4 v = crow[lane + i * 32];
            v.x *= scale; v.y *= scale; v.z *= scale; v.w *= scale;
            orow[lane + i * 32] = v;
        }
    }
}

// ---------------------------------------------------------------------------
// Launch: workspace = [256 f32 c_sq | pad to 1KB | 512KB bf16 packed codebook]
// ---------------------------------------------------------------------------
extern "C" void kernel_launch(void* const* d_in, const int* in_sizes, int n_in,
                              void* d_out, int out_size, void* d_ws, size_t ws_size,
                              hipStream_t stream) {
    const float* keys   = (const float*)d_in[0];
    const float* values = (const float*)d_in[1];
    const float* cb     = (const float*)d_in[2];
    float* out = (float*)d_out;

    float* c_sq = (float*)d_ws;
    unsigned short* cb_frag = (unsigned short*)((char*)d_ws + 1024);

    csq_kernel<<<1, 256, 0, stream>>>(cb, c_sq);
    pack_kernel<<<(NCODE * HDIM) / 256, 256, 0, stream>>>(cb, cb_frag);
    vq_main<<<(2 * NTILES_PER_TENSOR) / 8, 256, 0, stream>>>(
        keys, values, cb, c_sq, cb_frag, out);
}
